// ShfitScaleFormer_27350351741664
// MI455X (gfx1250) — compile-verified, matrix operand order checked by reading, not run on hardware
//
#include <hip/hip_runtime.h>

// ---------------- types ----------------
typedef __bf16 bf16_t;
typedef __attribute__((ext_vector_type(16))) __bf16 v16bf;
typedef __attribute__((ext_vector_type(8)))  __bf16 v8bf;
typedef __attribute__((ext_vector_type(8)))  float  v8f;

#define DEPTH   12
#define HEADS   12
#define DIMV    768
#define HD      64
#define NTOK    196
#define BATCH   32
#define TROWS   (BATCH*NTOK)     // 6272
#define RPB_TAB (7*13*13)        // 1183
#define SS_LD   208              // f32 score LDS stride / padded bias stride
#define SP_LD   224              // bf16 prob LDS stride (K padded to 224)
#define ATTN_LDS_BYTES (SS_LD*SS_LD*4 + SS_LD*SP_LD*2)   // 266,240 B

// ---------------- fragment helpers ----------------
// Load a 16x32 bf16 fragment (A layout; same pattern for B stored [n][k]).
// CDNA5 ISA 7.12.2: lanes 0-15 K-base 0, lanes 16-31 K-base 8,
// VGPRs 0-3 hold K..K+7, VGPRs 4-7 hold K+16..K+23.  Branchless.
__device__ __forceinline__ v16bf load_frag(const bf16_t* __restrict__ rowptr,
                                           int kk, int lane) {
  const int kb = (lane & 16) ? 8 : 0;
  v8bf lo = *(const v8bf*)(rowptr + kk + kb);
  v8bf hi = *(const v8bf*)(rowptr + kk + kb + 16);
  v16bf r;
#pragma unroll
  for (int i = 0; i < 8; ++i) { r[i] = lo[i]; r[i + 8] = hi[i]; }
  return r;
}

__device__ __forceinline__ v8f zero_acc() {
  v8f a;
#pragma unroll
  for (int i = 0; i < 8; ++i) a[i] = 0.0f;
  return a;
}

__device__ __forceinline__ float gelu_exact(float x) {
  return 0.5f * x * (1.0f + erff(x * 0.70710678118654752f));
}

// ---------------- generic WMMA GEMM ----------------
// C[M,N] = A[M,K](bf16) * Bt[N,K](bf16)^T + bias
// Wave tile 32x64 (2x4 sub-tiles, 8 WMMA per K-step), block tile 64x256.
// Out-of-range rows are CLAMPED (loads stay in-bounds, garbage rows are
// discarded by the m<M store guard) -> branchless inner loop.
// EPI: 0 = f32 store (+bias)         1 = bf16 store (+bias)
//      2 = bf16 store gelu(+bias)    3 = f32 store (+bias + residual)
// remap!=0: out_row = (m/49)*196 + row_add + m%49   (patch-embed concat)
template<int EPI>
__global__ __launch_bounds__(256) void gemm_wmma(
    const bf16_t* __restrict__ A, const bf16_t* __restrict__ Bt,
    const float* __restrict__ bias, const float* __restrict__ res,
    void* __restrict__ outv, int M, int N, int K, int remap, int row_add)
{
  const int lane = threadIdx.x & 31;
  const int wv   = threadIdx.x >> 5;      // 0..7
  const int wm   = wv >> 2;               // 0..1
  const int wn   = wv & 3;                // 0..3
  const int m0 = blockIdx.x * 64  + wm * 32;
  const int n0 = blockIdx.y * 256 + wn * 64;
  const int lr = lane & 15;

  int ra0 = m0 + lr;        if (ra0 >= M) ra0 = M - 1;
  int ra1 = m0 + 16 + lr;   if (ra1 >= M) ra1 = M - 1;
  const bf16_t* a0p = A + (size_t)ra0 * K;
  const bf16_t* a1p = A + (size_t)ra1 * K;
  const bf16_t* bp[4];
#pragma unroll
  for (int j = 0; j < 4; ++j) {
    int rb = n0 + j * 16 + lr;  if (rb >= N) rb = N - 1;
    bp[j] = Bt + (size_t)rb * K;
  }

  v8f acc[2][4];
#pragma unroll
  for (int i = 0; i < 2; ++i)
#pragma unroll
    for (int j = 0; j < 4; ++j) acc[i][j] = zero_acc();

#pragma unroll 2
  for (int kk = 0; kk < K; kk += 32) {
    v16bf a0 = load_frag(a0p, kk, lane);
    v16bf a1 = load_frag(a1p, kk, lane);
#pragma unroll
    for (int j = 0; j < 4; ++j) {
      v16bf b = load_frag(bp[j], kk, lane);
      acc[0][j] = __builtin_amdgcn_wmma_f32_16x16x32_bf16(false, a0, false, b, (short)0, acc[0][j], false, false);
      acc[1][j] = __builtin_amdgcn_wmma_f32_16x16x32_bf16(false, a1, false, b, (short)0, acc[1][j], false, false);
    }
  }

  const int mh = (lane & 16) ? 8 : 0;  // C/D layout: lanes 16-31 are M+8
#pragma unroll
  for (int j = 0; j < 4; ++j) {
    const int n = n0 + j * 16 + lr;
    if (n >= N) continue;
    const float bv = bias[n];
#pragma unroll
    for (int i = 0; i < 2; ++i) {
#pragma unroll
      for (int r = 0; r < 8; ++r) {
        int m = m0 + i * 16 + r + mh;
        if (m < M) {
          float v = acc[i][j][r] + bv;
          size_t orow = (size_t)m;
          if (remap) orow = (size_t)(m / 49) * 196 + row_add + (m % 49);
          if (EPI == 0)      ((float*)outv)[orow * N + n] = v;
          else if (EPI == 1) ((bf16_t*)outv)[orow * N + n] = (bf16_t)v;
          else if (EPI == 2) ((bf16_t*)outv)[orow * N + n] = (bf16_t)gelu_exact(v);
          else { v += res[orow * N + n]; ((float*)outv)[orow * N + n] = v; }
        }
      }
    }
  }
}

// ---------------- attention (one block per batch*head) ----------------
// bias_l: padded per-layer bias table [HEADS][208][208] f32 (pad = -1e30)
__global__ __launch_bounds__(256) void attn_kernel(
    const bf16_t* __restrict__ qh, const bf16_t* __restrict__ kh,
    const bf16_t* __restrict__ vt, const float* __restrict__ bias_l,
    bf16_t* __restrict__ o)
{
  extern __shared__ char smem[];
  float*  s_s = (float*)smem;                         // [208][208] scores f32
  bf16_t* s_p = (bf16_t*)(smem + SS_LD * SS_LD * 4);  // [208][224] probs bf16

  const int bh = blockIdx.x;
  const int b = bh / HEADS, h = bh % HEADS;
  const bf16_t* q = qh + (size_t)bh * NTOK * HD;   // q pre-scaled by 1/8
  const bf16_t* k = kh + (size_t)bh * NTOK * HD;
  const bf16_t* v = vt + (size_t)bh * HD * SP_LD;  // [64][224] (V^T, zero-padded)
  const float* bh_bias = bias_l + (size_t)h * SS_LD * SS_LD;

  const int lane = threadIdx.x & 31;
  const int wv   = threadIdx.x >> 5;
  const int lr   = lane & 15;
  const int mh   = (lane & 16) ? 8 : 0;

  // ---- S = q k^T + relpos bias : 13x13 tiles of 16x16, K=64 ----
  for (int t = wv; t < 169; t += 8) {
    const int mt = t / 13, nt = t % 13;
    int ra = mt * 16 + lr;  if (ra >= NTOK) ra = NTOK - 1;   // clamp, garbage masked by bias pad
    int rb = nt * 16 + lr;  if (rb >= NTOK) rb = NTOK - 1;
    const bf16_t* ap = q + (size_t)ra * HD;
    const bf16_t* bp = k + (size_t)rb * HD;
    v8f acc = zero_acc();
#pragma unroll
    for (int kk = 0; kk < HD; kk += 32) {
      v16bf a  = load_frag(ap, kk, lane);
      v16bf bb = load_frag(bp, kk, lane);
      acc = __builtin_amdgcn_wmma_f32_16x16x32_bf16(false, a, false, bb, (short)0, acc, false, false);
    }
#pragma unroll
    for (int r = 0; r < 8; ++r) {
      const int m = mt * 16 + r + mh;   // query index
      const int n = nt * 16 + lr;       // key index
      s_s[m * SS_LD + n] = acc[r] + bh_bias[m * SS_LD + n];
    }
  }
  __syncthreads();

  // ---- softmax over keys, repack to bf16 P (zero-padded to 208x224) ----
  const int tid = threadIdx.x;
  if (tid < NTOK) {
    float mx = -1e30f;
    for (int n = 0; n < NTOK; ++n) mx = fmaxf(mx, s_s[tid * SS_LD + n]);
    float sum = 0.0f;
    for (int n = 0; n < NTOK; ++n) {
      float e = __expf(s_s[tid * SS_LD + n] - mx);
      s_s[tid * SS_LD + n] = e;
      sum += e;
    }
    float inv = 1.0f / sum;
    for (int n = 0; n < NTOK; ++n)
      s_p[tid * SP_LD + n] = (bf16_t)(s_s[tid * SS_LD + n] * inv);
    for (int n = NTOK; n < SP_LD; ++n) s_p[tid * SP_LD + n] = (bf16_t)0.0f;
  } else if (tid < SS_LD) {
    for (int n = 0; n < SP_LD; ++n) s_p[tid * SP_LD + n] = (bf16_t)0.0f;
  }
  __syncthreads();

  // ---- O = P V : 13x4 tiles, K=224 (zero-padded) ----
  for (int t = wv; t < 52; t += 8) {
    const int mt = t / 4, nt = t % 4;
    const bf16_t* ap = s_p + (size_t)(mt * 16 + lr) * SP_LD;
    const bf16_t* bp = v   + (size_t)(nt * 16 + lr) * SP_LD;
    v8f acc = zero_acc();
#pragma unroll
    for (int kk = 0; kk < SP_LD; kk += 32) {
      v16bf a  = load_frag(ap, kk, lane);
      v16bf bb = load_frag(bp, kk, lane);
      acc = __builtin_amdgcn_wmma_f32_16x16x32_bf16(false, a, false, bb, (short)0, acc, false, false);
    }
#pragma unroll
    for (int r = 0; r < 8; ++r) {
      const int m = mt * 16 + r + mh;
      const int d = nt * 16 + lr;
      if (m < NTOK)
        o[((size_t)(b * NTOK + m)) * DIMV + h * HD + d] = (bf16_t)acc[r];
    }
  }
}

// ---------------- LayerNorm (one block per row) ----------------
template<int BF16OUT>
__global__ __launch_bounds__(256) void ln_kernel(
    const float* __restrict__ x, const float* __restrict__ g,
    const float* __restrict__ bb, void* __restrict__ outv,
    int D, int ostride, int ooff)
{
  __shared__ float s1[256], s2[256];
  const int row = blockIdx.x, tid = threadIdx.x;
  const float* xr = x + (size_t)row * D;
  float a = 0.0f, sq = 0.0f;
  for (int i = tid; i < D; i += 256) { float v = xr[i]; a += v; sq += v * v; }
  s1[tid] = a; s2[tid] = sq;
  __syncthreads();
  for (int s = 128; s > 0; s >>= 1) {
    if (tid < s) { s1[tid] += s1[tid + s]; s2[tid] += s2[tid + s]; }
    __syncthreads();
  }
  float mean = s1[0] / D;
  float var  = s2[0] / D - mean * mean;
  float rstd = rsqrtf(var + 1e-5f);
  for (int i = tid; i < D; i += 256) {
    float v = g[i] * (xr[i] - mean) * rstd + bb[i];
    if (BF16OUT) ((bf16_t*)outv)[(size_t)row * D + i] = (bf16_t)v;
    else         ((float*)outv)[(size_t)row * ostride + ooff + i] = v;
  }
}

// ---------------- small helper kernels ----------------
__global__ void relidx_kernel(int* __restrict__ relidx) {
  int idx = blockIdx.x * blockDim.x + threadIdx.x;
  if (idx >= NTOK * NTOK) return;
  int i = idx / NTOK, j = idx % NTOK;
  int c0i = i / 49, c1i = (i % 49) / 7, c2i = i % 7;
  int c0j = j / 49, c1j = (j % 49) / 7, c2j = j % 7;
  relidx[idx] = (c0i - c0j + 3) * 169 + (c1i - c1j + 6) * 13 + (c2i - c2j + 6);
}

// Expand rpb to padded dense bias: out[l][h][208][208]; pad cells = -1e30
__global__ void bias_expand_kernel(const float* __restrict__ rpb,
                                   const int* __restrict__ relidx,
                                   float* __restrict__ out) {
  long idx = (long)blockIdx.x * blockDim.x + threadIdx.x;
  if (idx >= (long)DEPTH * HEADS * SS_LD * SS_LD) return;
  int n = (int)(idx % SS_LD); long r = idx / SS_LD;
  int m = (int)(r % SS_LD); r /= SS_LD;
  int h = (int)(r % HEADS); int l = (int)(r / HEADS);
  float v = -1e30f;
  if (m < NTOK && n < NTOK)
    v = rpb[(size_t)l * RPB_TAB * HEADS + (size_t)relidx[m * NTOK + n] * HEADS + h];
  out[idx] = v;
}

// w[K][N] fp32 -> wt[N][Kp] bf16 (transpose, zero-pad K)
__global__ void wtrans_kernel(const float* __restrict__ w, bf16_t* __restrict__ wt,
                              int K, int N, int Kp) {
  long idx = (long)blockIdx.x * blockDim.x + threadIdx.x;
  if (idx >= (long)N * Kp) return;
  int kq = (int)(idx % Kp); long n = idx / Kp;
  wt[idx] = (kq < K) ? (bf16_t)w[(size_t)kq * N + n] : (bf16_t)0.0f;
}

// w[N][K] fp32 -> wt[N][Kp] bf16 (copy, zero-pad K)  -- patch-embed weights
__global__ void wcopy_pad_kernel(const float* __restrict__ w, bf16_t* __restrict__ wt,
                                 int N, int K, int Kp) {
  long idx = (long)blockIdx.x * blockDim.x + threadIdx.x;
  if (idx >= (long)N * Kp) return;
  int kq = (int)(idx % Kp); long n = idx / Kp;
  wt[idx] = (kq < K) ? (bf16_t)w[(size_t)n * K + kq] : (bf16_t)0.0f;
}

// img[B][3][H][H] -> im2col bf16 [B*49][Kp], Kp zero-padded
__global__ void im2col_kernel(const float* __restrict__ img, bf16_t* __restrict__ out,
                              int p, int K, int Kp) {
  long idx = (long)blockIdx.x * blockDim.x + threadIdx.x;
  if (idx >= (long)BATCH * 49 * Kp) return;
  int kq = (int)(idx % Kp); long t = idx / Kp;
  int loc = (int)(t % 49); int b = (int)(t / 49);
  if (kq >= K) { out[idx] = (bf16_t)0.0f; return; }
  int pp = p * p;
  int c = kq / pp, r = kq % pp, py = r / p, px = r % p;
  int gy = loc / 7, gx = loc % 7, H = p * 7;
  out[idx] = (bf16_t)img[(((size_t)b * 3 + c) * H + gy * p + py) * H + gx * p + px];
}

// qkv [T][2304] bf16 -> q (scaled) / k head-major [B][H][196][64]
__global__ void split_qk_kernel(const bf16_t* __restrict__ qkv,
                                bf16_t* __restrict__ qh, bf16_t* __restrict__ kh) {
  long idx = (long)blockIdx.x * blockDim.x + threadIdx.x;
  if (idx >= (long)BATCH * HEADS * NTOK * HD) return;
  int d = (int)(idx % HD); long r = idx / HD;
  int n = (int)(r % NTOK); r /= NTOK;
  int h = (int)(r % HEADS); int b = (int)(r / HEADS);
  size_t base = ((size_t)(b * NTOK + n)) * 2304 + h * HD + d;
  qh[idx] = (bf16_t)(0.125f * (float)qkv[base]);          // HD^-0.5
  kh[idx] = qkv[base + DIMV];
}

// v -> transposed [B][H][64][224] with zero padding for m in [196,224)
__global__ void split_vt_kernel(const bf16_t* __restrict__ qkv, bf16_t* __restrict__ vt) {
  long idx = (long)blockIdx.x * blockDim.x + threadIdx.x;
  if (idx >= (long)BATCH * HEADS * HD * SP_LD) return;
  int m = (int)(idx % SP_LD); long r = idx / SP_LD;
  int d = (int)(r % HD); r /= HD;
  int h = (int)(r % HEADS); int b = (int)(r / HEADS);
  vt[idx] = (m < NTOK)
      ? qkv[((size_t)(b * NTOK + m)) * 2304 + 2 * DIMV + h * HD + d]
      : (bf16_t)0.0f;
}

// designed[B][19] @ fe0_w(768,19)^T + b, gelu
__global__ void fmlp19_kernel(const float* __restrict__ f, const float* __restrict__ w,
                              const float* __restrict__ b, float* __restrict__ out) {
  int idx = blockIdx.x * blockDim.x + threadIdx.x;
  if (idx >= BATCH * DIMV) return;
  int n = idx % DIMV, bi = idx / DIMV;
  float s = b[n];
  for (int kq = 0; kq < 19; ++kq) s += f[bi * 19 + kq] * w[n * 19 + kq];
  out[idx] = gelu_exact(s);
}

// f[B][768] @ w(768,768)^T + b
__global__ void fmlp768_kernel(const float* __restrict__ f, const float* __restrict__ w,
                               const float* __restrict__ b, float* __restrict__ out) {
  int idx = blockIdx.x * blockDim.x + threadIdx.x;
  if (idx >= BATCH * DIMV) return;
  int n = idx % DIMV, bi = idx / DIMV;
  float s = b[n];
  for (int kq = 0; kq < DIMV; ++kq) s += f[(size_t)bi * DIMV + kq] * w[(size_t)n * DIMV + kq];
  out[idx] = s;
}

// mean-pool 4 groups of 49 tokens -> out[b][i*768+d]
__global__ void pool_kernel(const float* __restrict__ xn, float* __restrict__ out) {
  int idx = blockIdx.x * blockDim.x + threadIdx.x;
  if (idx >= BATCH * 4 * DIMV) return;
  int d = idx % DIMV; int r = idx / DIMV;
  int i = r % 4; int b = r / 4;
  float s = 0.0f;
  for (int t = 0; t < 49; ++t)
    s += xn[((size_t)(b * NTOK + i * 49 + t)) * DIMV + d];
  out[(size_t)b * 3840 + i * DIMV + d] = s * (1.0f / 49.0f);
}

// ---------------- host orchestration ----------------
extern "C" void kernel_launch(void* const* d_in, const int* in_sizes, int n_in,
                              void* d_out, int out_size, void* d_ws, size_t ws_size,
                              hipStream_t stream)
{
  (void)in_sizes; (void)n_in; (void)out_size; (void)ws_size;

  const float* img[4]   = {(const float*)d_in[0], (const float*)d_in[1],
                           (const float*)d_in[2], (const float*)d_in[3]};
  const float* designed = (const float*)d_in[4];
  const float* pe_w[4]  = {(const float*)d_in[5], (const float*)d_in[7],
                           (const float*)d_in[9], (const float*)d_in[11]};
  const float* pe_b[4]  = {(const float*)d_in[6], (const float*)d_in[8],
                           (const float*)d_in[10], (const float*)d_in[12]};
  const float* ln1_g = (const float*)d_in[13];
  const float* ln1_b = (const float*)d_in[14];
  const float* qkv_w = (const float*)d_in[15];
  const float* qkv_b = (const float*)d_in[16];
  const float* rpb   = (const float*)d_in[17];
  const float* proj_w = (const float*)d_in[18];
  const float* proj_b = (const float*)d_in[19];
  const float* ln2_g = (const float*)d_in[20];
  const float* ln2_b = (const float*)d_in[21];
  const float* fc1_w = (const float*)d_in[22];
  const float* fc1_b = (const float*)d_in[23];
  const float* fc2_w = (const float*)d_in[24];
  const float* fc2_b = (const float*)d_in[25];
  const float* norm_g = (const float*)d_in[26];
  const float* norm_b = (const float*)d_in[27];
  const float* fe0_w = (const float*)d_in[28];
  const float* fe0_b = (const float*)d_in[29];
  const float* fe1_w = (const float*)d_in[30];
  const float* fe1_b = (const float*)d_in[31];
  const float* fe2_w = (const float*)d_in[32];
  const float* fe2_b = (const float*)d_in[33];

  char* ws = (char*)d_ws;
  size_t off = 0;
  auto alloc = [&](size_t bytes) -> char* {
    char* p = ws + off;
    off = (off + bytes + 255) & ~(size_t)255;
    return p;
  };
  float*  x    = (float*) alloc((size_t)TROWS * DIMV * 4);
  bf16_t* h16  = (bf16_t*)alloc((size_t)TROWS * DIMV * 2);
  bf16_t* m16  = (bf16_t*)alloc((size_t)TROWS * 3072 * 2);   // also reused as xn f32
  bf16_t* qkvb = (bf16_t*)alloc((size_t)TROWS * 2304 * 2);
  bf16_t* qh   = (bf16_t*)alloc((size_t)BATCH * HEADS * NTOK * HD * 2);
  bf16_t* kh   = (bf16_t*)alloc((size_t)BATCH * HEADS * NTOK * HD * 2);
  bf16_t* vtb  = (bf16_t*)alloc((size_t)BATCH * HEADS * HD * SP_LD * 2);
  bf16_t* o16  = (bf16_t*)alloc((size_t)TROWS * DIMV * 2);
  bf16_t* wt   = (bf16_t*)alloc((size_t)3072 * 768 * 2);
  bf16_t* imc  = (bf16_t*)alloc((size_t)1568 * 3072 * 2);
  int*    rel  = (int*)   alloc((size_t)NTOK * NTOK * 4);
  float*  biasf = (float*)alloc((size_t)DEPTH * HEADS * SS_LD * SS_LD * 4); // 24.9 MB
  float*  f0   = (float*) alloc((size_t)BATCH * DIMV * 4);
  float*  f1   = (float*) alloc((size_t)BATCH * DIMV * 4);

  auto nb = [](long n) { return (int)((n + 255) / 256); };

  relidx_kernel<<<nb((long)NTOK * NTOK), 256, 0, stream>>>(rel);
  bias_expand_kernel<<<nb((long)DEPTH * HEADS * SS_LD * SS_LD), 256, 0, stream>>>(
      rpb, rel, biasf);

  // ---- patch embeds ----
  for (int e = 0; e < 4; ++e) {
    int p = 4 << e;
    int K = 3 * p * p;
    int Kp = (K < 64) ? 64 : K;
    wcopy_pad_kernel<<<nb((long)DIMV * Kp), 256, 0, stream>>>(pe_w[e], wt, DIMV, K, Kp);
    im2col_kernel<<<nb((long)BATCH * 49 * Kp), 256, 0, stream>>>(img[e], imc, p, K, Kp);
    dim3 g((1568 + 63) / 64, DIMV / 256);
    gemm_wmma<0><<<g, 256, 0, stream>>>(imc, wt, pe_b[e], nullptr, x,
                                        1568, DIMV, Kp, 1, e * 49);
  }

  const dim3 g_qkv(TROWS / 64, 2304 / 256);
  const dim3 g_768(TROWS / 64, DIMV / 256);
  const dim3 g_fc1(TROWS / 64, 3072 / 256);

  // ---- transformer layers ----
  for (int l = 0; l < DEPTH; ++l) {
    ln_kernel<1><<<TROWS, 256, 0, stream>>>(x, ln1_g + l * DIMV, ln1_b + l * DIMV,
                                            h16, DIMV, 0, 0);
    wtrans_kernel<<<nb((long)2304 * 768), 256, 0, stream>>>(
        qkv_w + (size_t)l * DIMV * 2304, wt, DIMV, 2304, DIMV);
    gemm_wmma<1><<<g_qkv, 256, 0, stream>>>(h16, wt, qkv_b + (size_t)l * 2304,
                                            nullptr, qkvb, TROWS, 2304, DIMV, 0, 0);
    split_qk_kernel<<<nb((long)BATCH * HEADS * NTOK * HD), 256, 0, stream>>>(qkvb, qh, kh);
    split_vt_kernel<<<nb((long)BATCH * HEADS * HD * SP_LD), 256, 0, stream>>>(qkvb, vtb);
    attn_kernel<<<BATCH * HEADS, 256, ATTN_LDS_BYTES, stream>>>(
        qh, kh, vtb, biasf + (size_t)l * HEADS * SS_LD * SS_LD, o16);
    wtrans_kernel<<<nb((long)768 * 768), 256, 0, stream>>>(
        proj_w + (size_t)l * DIMV * DIMV, wt, DIMV, DIMV, DIMV);
    gemm_wmma<3><<<g_768, 256, 0, stream>>>(o16, wt, proj_b + (size_t)l * DIMV,
                                            x, x, TROWS, DIMV, DIMV, 0, 0);
    ln_kernel<1><<<TROWS, 256, 0, stream>>>(x, ln2_g + l * DIMV, ln2_b + l * DIMV,
                                            h16, DIMV, 0, 0);
    wtrans_kernel<<<nb((long)3072 * 768), 256, 0, stream>>>(
        fc1_w + (size_t)l * DIMV * 3072, wt, DIMV, 3072, DIMV);
    gemm_wmma<2><<<g_fc1, 256, 0, stream>>>(h16, wt, fc1_b + (size_t)l * 3072,
                                            nullptr, m16, TROWS, 3072, DIMV, 0, 0);
    wtrans_kernel<<<nb((long)768 * 3072), 256, 0, stream>>>(
        fc2_w + (size_t)l * 3072 * DIMV, wt, 3072, DIMV, 3072);
    gemm_wmma<3><<<g_768, 256, 0, stream>>>(m16, wt, fc2_b + (size_t)l * DIMV,
                                            x, x, TROWS, DIMV, 3072, 0, 0);
  }

  // ---- final norm + pools ----
  float* xn = (float*)m16;   // reuse MLP scratch (38.5MB >= 19.3MB)
  ln_kernel<0><<<TROWS, 256, 0, stream>>>(x, norm_g, norm_b, xn, DIMV, DIMV, 0);
  pool_kernel<<<nb((long)BATCH * 4 * DIMV), 256, 0, stream>>>(xn, (float*)d_out);

  // ---- designed-feature MLP ----
  fmlp19_kernel<<<nb((long)BATCH * DIMV), 256, 0, stream>>>(designed, fe0_w, fe0_b, f0);
  fmlp768_kernel<<<nb((long)BATCH * DIMV), 256, 0, stream>>>(f0, fe1_w, fe1_b, f1);
  fmlp768_kernel<<<nb((long)BATCH * DIMV), 256, 0, stream>>>(f1, fe2_w, fe2_b, f0);
  ln_kernel<0><<<BATCH, 256, 0, stream>>>(f0, norm_g, norm_b, (float*)d_out,
                                          DIMV, 3840, 4 * DIMV);
}
